// MQA_1357209666294
// MI455X (gfx1250) — matrix operand order repaired and checked by reading, not compiled, not run on hardware
//
#include <hip/hip_runtime.h>
#include <hip/hip_bf16.h>

// ---------- CDNA5 WMMA types ----------
typedef __attribute__((ext_vector_type(16))) __bf16 v16bf;
typedef __attribute__((ext_vector_type(8)))  __bf16 v8bf;
typedef __attribute__((ext_vector_type(4)))  __bf16 v4bf;
typedef __attribute__((ext_vector_type(8)))  float  v8f;

__device__ __forceinline__ __bf16 f2bf(float x) { return (__bf16)x; }
__device__ __forceinline__ float  bf2f(__bf16 x) { return (float)x; }

// Problem constants (reference setup_inputs is fixed).
#define CB   2
#define CS   2048
#define CD   2048
#define CH   16
#define CKD  128
#define CNQ  2304   // D + 2*kd

// =====================================================================
// Kernel 1: C[M,N] = A[M,K] @ W[K,N] + bias[N]   (bf16 WMMA, f32 accum)
// BM=BN=128: balances A/W L2 streaming traffic (~0.5 GB each) and doubles
// arithmetic intensity vs 64x128. Double-buffered LDS; global loads for
// step k+1 overlap the 8 WMMAs of step k; one barrier per K-step.
// Wave w: wm2 = w&1 -> 64-row half (4 M-tiles); wn2 = w>>1 -> 32-col slab
// (2 N-tiles); 8 accumulator tiles per wave.
// =====================================================================
#define BM 128
#define BN 128
#define BK 32
#define APAD 40   // row stride (elements) -> 80B; 8B/16B aligned sub-chunks

__device__ __forceinline__ void stage_a(__bf16 (*As)[APAD], int r, int c, float4 a) {
  v4bf p; p[0] = f2bf(a.x); p[1] = f2bf(a.y); p[2] = f2bf(a.z); p[3] = f2bf(a.w);
  *(v4bf*)&As[r][c] = p;
}
__device__ __forceinline__ void stage_bt(__bf16 (*Bt)[APAD], int n, int kk, float4 b) {
  Bt[n + 0][kk] = f2bf(b.x);
  Bt[n + 1][kk] = f2bf(b.y);
  Bt[n + 2][kk] = f2bf(b.z);
  Bt[n + 3][kk] = f2bf(b.w);
}

template <bool BF16OUT>
__global__ __launch_bounds__(256)
void gemm_bias_bf16(const float* __restrict__ A, const float* __restrict__ W,
                    const float* __restrict__ bias, void* __restrict__ Cout,
                    int M, int N, int K) {
  __shared__ __bf16 As[2][BM][APAD];   // A tile, row-major (M x K)
  __shared__ __bf16 Bt[2][BN][APAD];   // W tile, transposed (N x K)

  const int tid  = threadIdx.x;
  const int w    = tid >> 5;
  const int lane = tid & 31;
  const int half = lane >> 4;
  const int l16  = lane & 15;
  const int m0   = blockIdx.x * BM;
  const int n0   = blockIdx.y * BN;
  const int wm2  = w & 1;    // 64-row half
  const int wn2  = w >> 1;   // 32-col slab

  // Staging coordinates (float4 granularity). A: 1024 quads, B: 1024 quads.
  int arr[4], acc_[4], bkk[4], bnn[4];
  #pragma unroll
  for (int i = 0; i < 4; ++i) {
    int q = tid + 256 * i;
    arr[i]  = q >> 3;        // A row
    acc_[i] = (q & 7) * 4;   // A col
    bkk[i]  = q >> 5;        // W row (k)
    bnn[i]  = (q & 31) * 4;  // W col (n)
  }

  v8f acc[4][2] = {};

  // ---- prologue: load + stage K-step 0 ----
  {
    float4 av[4], bw[4];
    #pragma unroll
    for (int i = 0; i < 4; ++i)
      av[i] = *(const float4*)&A[(size_t)(m0 + arr[i]) * K + acc_[i]];
    #pragma unroll
    for (int i = 0; i < 4; ++i)
      bw[i] = *(const float4*)&W[(size_t)bkk[i] * N + (n0 + bnn[i])];
    #pragma unroll
    for (int i = 0; i < 4; ++i) stage_a(As[0], arr[i], acc_[i], av[i]);
    #pragma unroll
    for (int i = 0; i < 4; ++i) stage_bt(Bt[0], bnn[i], bkk[i], bw[i]);
  }
  __syncthreads();

  int buf = 0;
  for (int k0 = 0; k0 < K; k0 += BK) {
    const bool has_next = (k0 + BK) < K;
    float4 av[4], bw[4];
    if (has_next) {   // issue next step's global loads early
      #pragma unroll
      for (int i = 0; i < 4; ++i)
        av[i] = *(const float4*)&A[(size_t)(m0 + arr[i]) * K + (k0 + BK + acc_[i])];
      #pragma unroll
      for (int i = 0; i < 4; ++i)
        bw[i] = *(const float4*)&W[(size_t)(k0 + BK + bkk[i]) * N + (n0 + bnn[i])];
    }

    // ---- compute current step from LDS buf ----
    v16bf afr[4];
    #pragma unroll
    for (int mt = 0; mt < 4; ++mt) {
      const __bf16* ap = &As[buf][wm2 * 64 + mt * 16 + l16][0];
      v8bf lo = *(const v8bf*)&ap[16 * half];
      v8bf hi = *(const v8bf*)&ap[16 * half + 8];
      #pragma unroll
      for (int j = 0; j < 8; ++j) { afr[mt][j] = lo[j]; afr[mt][8 + j] = hi[j]; }
    }
    v16bf bfr[2];
    #pragma unroll
    for (int nt = 0; nt < 2; ++nt) {
      const __bf16* bp = &Bt[buf][wn2 * 32 + nt * 16 + l16][0];
      v8bf lo = *(const v8bf*)&bp[16 * half];
      v8bf hi = *(const v8bf*)&bp[16 * half + 8];
      #pragma unroll
      for (int j = 0; j < 8; ++j) { bfr[nt][j] = lo[j]; bfr[nt][8 + j] = hi[j]; }
    }
    #pragma unroll
    for (int mt = 0; mt < 4; ++mt)
      #pragma unroll
      for (int nt = 0; nt < 2; ++nt)
        acc[mt][nt] = __builtin_amdgcn_wmma_f32_16x16x32_bf16(
            false, afr[mt], false, bfr[nt], (short)0, acc[mt][nt], false, false);

    if (has_next) {   // stage next step into the other buffer
      #pragma unroll
      for (int i = 0; i < 4; ++i) stage_a(As[buf ^ 1], arr[i], acc_[i], av[i]);
      #pragma unroll
      for (int i = 0; i < 4; ++i) stage_bt(Bt[buf ^ 1], bnn[i], bkk[i], bw[i]);
    }
    __syncthreads();
    buf ^= 1;
  }

  // Epilogue: add bias; non-temporal stores for f32 output, regular stores for bf16.
  #pragma unroll
  for (int nt = 0; nt < 2; ++nt) {
    int col = n0 + wn2 * 32 + nt * 16 + l16;
    float bv = bias[col];
    #pragma unroll
    for (int mt = 0; mt < 4; ++mt) {
      #pragma unroll
      for (int r = 0; r < 8; ++r) {
        int row = m0 + wm2 * 64 + mt * 16 + r + 8 * half;
        float v = acc[mt][nt][r] + bv;
        if (BF16OUT)   // qkv workspace IS re-read: regular store
          ((__bf16*)Cout)[(size_t)row * N + col] = f2bf(v);
        else           // short_cut: never re-read -> bypass-ish NT store
          __builtin_nontemporal_store(v, &((float*)Cout)[(size_t)row * N + col]);
      }
    }
  }
}

// =====================================================================
// Kernel 2: fused score + softmax + P@V per (b, h, 32-row s-block).
// Score strip (32 x 2048) stays resident in LDS as bf16 (~128 KB):
// score is written to HBM exactly once (non-temporal) and never re-read.
// Phase 1 pipelines k-fragment loads (register double buffer); phase 2
// double-buffers the transposed v tile in LDS.
// =====================================================================
#define SLDP 2056   // padded row stride (4112B, 16B aligned)
#define VPAD 40

__global__ __launch_bounds__(256)
void mqa_attn_fused(const __bf16* __restrict__ qkv, float* __restrict__ score,
                    float* __restrict__ atten, float* __restrict__ res) {
  __shared__ __bf16 Pld[32][SLDP];      // score/prob strip (bf16)
  __shared__ __bf16 vT[2][CKD][VPAD];   // transposed v tiles (double buffer)
  __shared__ float  red[32][8];
  __shared__ float  rmax[32];
  __shared__ float  rsum[32];

  const int tid  = threadIdx.x;
  const int w    = tid >> 5;
  const int lane = tid & 31;
  const int half = lane >> 4;
  const int l16  = lane & 15;
  const int sb = blockIdx.x, h = blockIdx.y, b = blockIdx.z;
  const int s0 = sb * 32;

  const __bf16* qbase = qkv + (size_t)b * CS * CNQ + (size_t)h * CKD;
  const __bf16* kbase = qkv + (size_t)b * CS * CNQ + CD;
  const __bf16* vbase = qkv + (size_t)b * CS * CNQ + CD + CKD;

  // ---------------- Phase 1: score strip ----------------
  const int m = w & 1;                 // 16-row half of the strip
  const __bf16* qrow = qbase + (size_t)(s0 + m * 16 + l16) * CNQ;

  v16bf aq[4];   // q fragments for K = 0..127 in chunks of 32
  #pragma unroll
  for (int c = 0; c < 4; ++c) {
    v8bf lo = *(const v8bf*)&qrow[32 * c + 16 * half];
    v8bf hi = *(const v8bf*)&qrow[32 * c + 16 * half + 8];
    #pragma unroll
    for (int j = 0; j < 8; ++j) { aq[c][j] = lo[j]; aq[c][8 + j] = hi[j]; }
  }

  const int tt0 = (w >> 1) * 32;       // 32 t-tiles (of 16 cols) per wave
  const float inv_scale = 1.0f / 16384.0f;

  v16bf bk[4];   // preload k fragments for first tile
  {
    const __bf16* krow = kbase + (size_t)(tt0 * 16 + l16) * CNQ;
    #pragma unroll
    for (int c = 0; c < 4; ++c) bk[c] = *(const v16bf*)&krow[32 * c + 16 * half];
  }

  for (int ti = 0; ti < 32; ++ti) {
    int tt = tt0 + ti;
    v16bf bkn[4];
    if (ti < 31) {   // next tile's loads overlap this tile's WMMAs
      const __bf16* krn = kbase + (size_t)((tt + 1) * 16 + l16) * CNQ;
      __builtin_prefetch(krn + (size_t)16 * CNQ, 0, 3);
      #pragma unroll
      for (int c = 0; c < 4; ++c) bkn[c] = *(const v16bf*)&krn[32 * c + 16 * half];
    }
    v8f acc = {};
    #pragma unroll
    for (int c = 0; c < 4; ++c)
      acc = __builtin_amdgcn_wmma_f32_16x16x32_bf16(
          false, aq[c], false, bk[c], (short)0, acc, false, false);

    size_t sc_base = ((size_t)(b * CH + h) * CS + (s0 + m * 16)) * CS + (size_t)tt * 16 + l16;
    #pragma unroll
    for (int r = 0; r < 8; ++r) {
      int rr = r + 8 * half;
      float vsc = acc[r] * inv_scale;
      __builtin_nontemporal_store(vsc, &score[sc_base + (size_t)rr * CS]);
      Pld[m * 16 + rr][tt * 16 + l16] = f2bf(vsc); // LDS-resident copy
    }
    if (ti < 31) {
      #pragma unroll
      for (int c = 0; c < 4; ++c) bk[c] = bkn[c];
    }
  }
  __syncthreads();

  // ---------------- Phase 1.5: row softmax in LDS (vectorized) ----------------
  const int prow = tid >> 3, sub = tid & 7;   // 8 threads per row, 256 cols each
  {
    float mx = -1e30f;
    for (int c = sub * 256; c < sub * 256 + 256; c += 8) {
      v8bf v = *(const v8bf*)&Pld[prow][c];
      #pragma unroll
      for (int j = 0; j < 8; ++j) mx = fmaxf(mx, bf2f(v[j]));
    }
    red[prow][sub] = mx;
  }
  __syncthreads();
  if (tid < 32) {
    float mx = red[tid][0];
    #pragma unroll
    for (int i = 1; i < 8; ++i) mx = fmaxf(mx, red[tid][i]);
    rmax[tid] = mx;
  }
  __syncthreads();
  {
    float mx = rmax[prow], sm = 0.0f;
    for (int c = sub * 256; c < sub * 256 + 256; c += 8) {
      v8bf v = *(const v8bf*)&Pld[prow][c];
      #pragma unroll
      for (int j = 0; j < 8; ++j) sm += __expf(bf2f(v[j]) - mx);
    }
    red[prow][sub] = sm;
  }
  __syncthreads();
  if (tid < 32) {
    float sm = 0.0f;
    #pragma unroll
    for (int i = 0; i < 8; ++i) sm += red[tid][i];
    rsum[tid] = sm;
  }
  __syncthreads();
  {
    float mx = rmax[prow], inv = 1.0f / rsum[prow];
    for (int c = sub * 256; c < sub * 256 + 256; c += 8) {
      v8bf v = *(const v8bf*)&Pld[prow][c];
      v8bf o;
      #pragma unroll
      for (int j = 0; j < 8; ++j) o[j] = f2bf(__expf(bf2f(v[j]) - mx) * inv);
      *(v8bf*)&Pld[prow][c] = o;
    }
  }
  __syncthreads();

  // ---------------- Phase 2: atten = P @ v (double-buffered vT) ----------------
  const int nb = w >> 1;               // two 16-col d tiles per wave
  const int tl0 = tid >> 4,         vd0 = (tid & 15) * 8;
  const int tl1 = (tid + 256) >> 4, vd1 = ((tid + 256) & 15) * 8;
  v8f oacc[2] = {};

  {  // prologue: stage v tile 0
    v8bf g0 = *(const v8bf*)&vbase[(size_t)tl0 * CNQ + vd0];
    v8bf g1 = *(const v8bf*)&vbase[(size_t)tl1 * CNQ + vd1];
    #pragma unroll
    for (int j = 0; j < 8; ++j) {
      vT[0][vd0 + j][tl0] = g0[j];
      vT[0][vd1 + j][tl1] = g1[j];
    }
  }
  __syncthreads();

  int buf = 0;
  for (int kk = 0; kk < CS / 32; ++kk) {
    const bool has_next = kk < (CS / 32 - 1);
    v8bf g0, g1;
    if (has_next) {   // next tile's global loads overlap this tile's WMMAs
      int tn = (kk + 1) * 32;
      g0 = *(const v8bf*)&vbase[(size_t)(tn + tl0) * CNQ + vd0];
      g1 = *(const v8bf*)&vbase[(size_t)(tn + tl1) * CNQ + vd1];
    }

    int t0 = kk * 32;
    v16bf ap;   // A = P tile 16x32 from the LDS strip
    {
      const __bf16* pp = &Pld[m * 16 + l16][t0];
      v8bf lo = *(const v8bf*)&pp[16 * half];
      v8bf hi = *(const v8bf*)&pp[16 * half + 8];
      #pragma unroll
      for (int j = 0; j < 8; ++j) { ap[j] = lo[j]; ap[8 + j] = hi[j]; }
    }
    #pragma unroll
    for (int j2 = 0; j2 < 2; ++j2) {
      int ntile = nb * 2 + j2;
      v16bf bv;   // B = v tile 32x16: lane = d (N), contiguous t (K) per lane
      const __bf16* vp = &vT[buf][ntile * 16 + l16][0];
      v8bf lo = *(const v8bf*)&vp[16 * half];
      v8bf hi = *(const v8bf*)&vp[16 * half + 8];
      #pragma unroll
      for (int j = 0; j < 8; ++j) { bv[j] = lo[j]; bv[8 + j] = hi[j]; }
      oacc[j2] = __builtin_amdgcn_wmma_f32_16x16x32_bf16(
          false, ap, false, bv, (short)0, oacc[j2], false, false);
    }

    if (has_next) {
      #pragma unroll
      for (int j = 0; j < 8; ++j) {
        vT[buf ^ 1][vd0 + j][tl0] = g0[j];
        vT[buf ^ 1][vd1 + j][tl1] = g1[j];
      }
    }
    __syncthreads();
    buf ^= 1;
  }

  // Epilogue: write atten (b,h,s,d) and res (b,s,h*kd+d) — both streaming.
  #pragma unroll
  for (int j2 = 0; j2 < 2; ++j2) {
    int dcol = (nb * 2 + j2) * 16 + l16;
    #pragma unroll
    for (int r = 0; r < 8; ++r) {
      int s = s0 + m * 16 + r + 8 * half;
      float v = oacc[j2][r];
      __builtin_nontemporal_store(v, &atten[((size_t)(b * CH + h) * CS + s) * CKD + dcol]);
      __builtin_nontemporal_store(v, &res[((size_t)(b * CS + s)) * CD + h * CKD + dcol]);
    }
  }
}

// =====================================================================
extern "C" void kernel_launch(void* const* d_in, const int* in_sizes, int n_in,
                              void* d_out, int out_size, void* d_ws, size_t ws_size,
                              hipStream_t stream) {
  (void)in_sizes; (void)n_in; (void)out_size; (void)ws_size;
  const float* in_x  = (const float*)d_in[0];
  const float* W_sc  = (const float*)d_in[1];
  const float* b_sc  = (const float*)d_in[2];
  const float* W_qkv = (const float*)d_in[3];
  const float* b_qkv = (const float*)d_in[4];
  // d_in[5] = head_num (fixed at 16)

  float* out       = (float*)d_out;
  float* score     = out;                                         // 2*16*2048*2048
  float* atten     = out + (size_t)134217728;                     // 2*16*2048*128
  float* res       = out + (size_t)134217728 + 8388608;           // 2*2048*2048
  float* short_cut = out + (size_t)134217728 + 8388608 + 8388608; // 2*2048*2048
  __bf16* qkv_ws   = (__bf16*)d_ws;                               // 4096*2304 bf16

  const int M = CB * CS;   // 4096
  dim3 blk(256);

  // short_cut = in_x @ W_sc + b_sc   (f32 NT out: it is a kernel output)
  gemm_bias_bf16<false><<<dim3(M / BM, CD / BN), blk, 0, stream>>>(
      in_x, W_sc, b_sc, (void*)short_cut, M, CD, CD);
  // qkv = in_x @ W_qkv + b_qkv      (bf16 out: feeds bf16 WMMA attention)
  gemm_bias_bf16<true><<<dim3(M / BM, CNQ / BN), blk, 0, stream>>>(
      in_x, W_qkv, b_qkv, (void*)qkv_ws, M, CNQ, CD);
  // fused score + softmax + P@V (+ res transpose)
  mqa_attn_fused<<<dim3(CS / 32, CH, CB), blk, 0, stream>>>(
      qkv_ws, score, atten, res);
}
